// DarcyFlowDecoder_53034256171297
// MI455X (gfx1250) — matrix operand order repaired and checked by reading, not compiled, not run on hardware
//
#include <hip/hip_runtime.h>
#include <hip/hip_bf16.h>
#include <math.h>

// ---------------------------------------------------------------------------
// DarcyFlowDecoder on gfx1250 (MI455X).
// All large GEMMs go through v_wmma_f32_16x16x32_f16 (f16 A/B, fp32 accum),
// A tiles staged into LDS with global_load_async_to_lds_b128 (ASYNCcnt path).
// Workspace requirement: ~275 MB.
// ---------------------------------------------------------------------------

typedef _Float16 half_t;
typedef __attribute__((ext_vector_type(16))) _Float16 v16h;
typedef __attribute__((ext_vector_type(8)))  _Float16 v8h;
typedef __attribute__((ext_vector_type(8)))  float    v8f;

#define B_     4
#define NIN    4096
#define NQ     16384
#define NTOKQ  (B_ * NQ)     // 65536
#define NTOKIN (B_ * NIN)    // 16384
#define EPS_   1e-5f

__device__ __forceinline__ float gelu_f(float x) {
  // exact GELU: 0.5*x*(1+erf(x/sqrt(2)))
  return 0.5f * x * (1.0f + erff(x * 0.70710678118654752f));
}

// ---------------- elementwise / reduction kernels ----------------

__global__ void k_cvt_f32_f16(const float* __restrict__ src, half_t* __restrict__ dst, int n) {
  int i = blockIdx.x * blockDim.x + threadIdx.x;
  if (i < n) dst[i] = (half_t)src[i];
}

// Gaussian Fourier features: out[t, 0:64]=sin(2pi p@B), out[t, 64:128]=cos(...)
__global__ void k_fourier(const float* __restrict__ pos, const float* __restrict__ bff,
                          half_t* __restrict__ out, int ntok) {
  int i = blockIdx.x * blockDim.x + threadIdx.x;
  int t = i >> 6, j = i & 63;
  if (t >= ntok) return;
  float p0 = pos[(size_t)t * 2 + 0];
  float p1 = pos[(size_t)t * 2 + 1];
  float a = 6.28318530717958647f * (p0 * bff[j] + p1 * bff[64 + j]);
  out[(size_t)t * 128 + j]      = (half_t)sinf(a);
  out[(size_t)t * 128 + 64 + j] = (half_t)cosf(a);
}

__device__ __forceinline__ float block_sum(float v, float* sh) {
  int t = threadIdx.x;
  sh[t] = v;
  __syncthreads();
  for (int s = blockDim.x >> 1; s > 0; s >>= 1) {
    if (t < s) sh[t] += sh[t + s];
    __syncthreads();
  }
  float r = sh[0];
  __syncthreads();
  return r;
}

// LayerNorm over 128-wide rows -> f16 (used for z = LN(h))
__global__ void k_ln128(const float* __restrict__ in, const float* __restrict__ g,
                        const float* __restrict__ b, half_t* __restrict__ outh) {
  __shared__ float sh[128];
  int r = blockIdx.x, t = threadIdx.x;
  float v = in[(size_t)r * 128 + t];
  float m = block_sum(v, sh) * (1.0f / 128.0f);
  float d = v - m;
  float var = block_sum(d * d, sh) * (1.0f / 128.0f);
  float o = d * rsqrtf(var + EPS_) * g[t] + b[t];
  outh[(size_t)r * 128 + t] = (half_t)o;
}

// x2 = LN(attn) + xres  -> f32 and f16 copies
__global__ void k_ln128_res(const float* __restrict__ attn, const float* __restrict__ xres,
                            const float* __restrict__ g, const float* __restrict__ b,
                            float* __restrict__ out32, half_t* __restrict__ out16) {
  __shared__ float sh[128];
  int r = blockIdx.x, t = threadIdx.x;
  size_t idx = (size_t)r * 128 + t;
  float v = attn[idx];
  float m = block_sum(v, sh) * (1.0f / 128.0f);
  float d = v - m;
  float var = block_sum(d * d, sh) * (1.0f / 128.0f);
  float o = d * rsqrtf(var + EPS_) * g[t] + b[t] + xres[idx];
  out32[idx] = o;
  out16[idx] = (half_t)o;
}

// hd = LN(ffn_out + x2)  -> f16
__global__ void k_addln128(const float* __restrict__ a, const float* __restrict__ c,
                           const float* __restrict__ g, const float* __restrict__ b,
                           half_t* __restrict__ outh) {
  __shared__ float sh[128];
  int r = blockIdx.x, t = threadIdx.x;
  size_t idx = (size_t)r * 128 + t;
  float v = a[idx] + c[idx];
  float m = block_sum(v, sh) * (1.0f / 128.0f);
  float d = v - m;
  float var = block_sum(d * d, sh) * (1.0f / 128.0f);
  float o = d * rsqrtf(var + EPS_) * g[t] + b[t];
  outh[idx] = (half_t)o;
}

// 2D RoPE. in: [tok, 512] f16. q variant writes packed [z=(b*4+h)][n][128].
// freqs: f[j] = coord * 64 * 10000^(-(j%32)/32), halves of head dim use x / y coords.
__device__ __forceinline__ float rope_val(const half_t* __restrict__ src, size_t tok_base,
                                          int hh, int j, float coord) {
  int halfsel = j >> 6;          // 0: x-half, 1: y-half
  int jj = j & 63;
  int i32 = jj & 31;
  float f = coord * 64.0f * powf(10000.0f, -(float)i32 / 32.0f);
  int partner = (halfsel << 6) + ((jj < 32) ? (jj + 32) : (jj - 32));
  float sgn = (jj < 32) ? -1.0f : 1.0f;
  float tv = (float)src[tok_base + (hh << 7) + j];
  float pv = (float)src[tok_base + (hh << 7) + partner];
  return tv * cosf(f) + sgn * pv * sinf(f);
}

__global__ void k_rope_q(const half_t* __restrict__ q, const float* __restrict__ pos,
                         half_t* __restrict__ qp) {
  int i = blockIdx.x * blockDim.x + threadIdx.x;   // NTOKQ * 512 threads
  int t = i >> 9;
  int dim = i & 511;
  if (t >= NTOKQ) return;
  int bb = t >> 14;           // / NQ
  int n = t & (NQ - 1);
  int hh = dim >> 7, j = dim & 127;
  float coord = pos[(size_t)t * 2 + (j >> 6)];
  float o = rope_val(q, (size_t)t * 512, hh, j, coord);
  qp[(((size_t)(bb * 4 + hh) * NQ + n) << 7) + j] = (half_t)o;
}

__global__ void k_rope_k(const half_t* __restrict__ k, const float* __restrict__ pos,
                         half_t* __restrict__ kr) {
  int i = blockIdx.x * blockDim.x + threadIdx.x;   // NTOKIN * 512 threads
  int t = i >> 9;
  int dim = i & 511;
  if (t >= NTOKIN) return;
  int hh = dim >> 7, j = dim & 127;
  float coord = pos[(size_t)t * 2 + (j >> 6)];
  float o = rope_val(k, (size_t)t * 512, hh, j, coord);
  kr[(size_t)t * 512 + dim] = (half_t)o;
}

// Instance norm over the sequence dim per (batch, feature). One block per column.
// k variant writes TRANSPOSED [z][d][n]; v variant writes packed [z][n][d].
__global__ void k_inorm_kT(const half_t* __restrict__ in, half_t* __restrict__ outT, int nin) {
  __shared__ float sh[256];
  int col = blockIdx.x;               // 0..(4*512-1)
  int bb = col >> 9, feat = col & 511;
  const half_t* base = in + (size_t)bb * nin * 512 + feat;
  float s = 0.f, ss = 0.f;
  for (int n = threadIdx.x; n < nin; n += 256) {
    float v = (float)base[(size_t)n * 512];
    s += v; ss += v * v;
  }
  s = block_sum(s, sh);
  ss = block_sum(ss, sh);
  float mean = s / (float)nin;
  float inv = rsqrtf(ss / (float)nin - mean * mean + EPS_);
  int hh = feat >> 7, d = feat & 127;
  half_t* out = outT + ((size_t)(bb * 4 + hh) * 128 + d) * (size_t)nin;
  for (int n = threadIdx.x; n < nin; n += 256)
    out[n] = (half_t)(((float)base[(size_t)n * 512] - mean) * inv);
}

__global__ void k_inorm_v(const half_t* __restrict__ in, half_t* __restrict__ outp, int nin) {
  __shared__ float sh[256];
  int col = blockIdx.x;
  int bb = col >> 9, feat = col & 511;
  const half_t* base = in + (size_t)bb * nin * 512 + feat;
  float s = 0.f, ss = 0.f;
  for (int n = threadIdx.x; n < nin; n += 256) {
    float v = (float)base[(size_t)n * 512];
    s += v; ss += v * v;
  }
  s = block_sum(s, sh);
  ss = block_sum(ss, sh);
  float mean = s / (float)nin;
  float inv = rsqrtf(ss / (float)nin - mean * mean + EPS_);
  int hh = feat >> 7, d = feat & 127;
  half_t* out = outp + ((size_t)(bb * 4 + hh) * nin) * 128 + d;
  for (int n = threadIdx.x; n < nin; n += 256)
    out[(size_t)n * 128] = (half_t)(((float)base[(size_t)n * 512] - mean) * inv);
}

// final: out[t] = d1[t,:] @ dec_w2   (64-long dot)
__global__ void k_decode(const half_t* __restrict__ d1, const float* __restrict__ w2,
                         float* __restrict__ out, int ntok) {
  int t = blockIdx.x * blockDim.x + threadIdx.x;
  if (t >= ntok) return;
  float acc = 0.f;
  const half_t* row = d1 + (size_t)t * 64;
#pragma unroll
  for (int j = 0; j < 64; ++j) acc += (float)row[j] * w2[j];
  out[t] = acc;
}

// ---------------- WMMA GEMM kernel ----------------
// C[M,N] = epilogue(A[M,K] * B[K,N]).  A,B f16 row-major; fp32 accum.
// Requires M%128==0, N%64==0, K%64==0 (all call sites conform).
// Batch z = blockIdx.z: per-matrix offset = (z>>2)*s?b + (z&3)*s?h.
// A tile staged into LDS via async DMA (global_load_async_to_lds_b128,
// ASYNCcnt-tracked); B tile staged transposed through VGPRs (ds scatter).
#define TM  128
#define TN  64
#define TKB 64   // K staged per LDS round -> 8 v_wmma per barrier pair

__global__ __launch_bounds__(256) void k_wmma_gemm(
    const half_t* __restrict__ A, const half_t* __restrict__ B,
    float* __restrict__ C32, half_t* __restrict__ C16,
    const float* __restrict__ bias,
    int M, int N, int K, int lda, int ldb, int ldc,
    long long sAb, long long sAh, long long sBb, long long sBh,
    long long sCb, long long sCh, float scale, int do_gelu) {
  __shared__ half_t As[TM][TKB + 8];   // 128 x 72 halves (row pitch 144B, 16B aligned)
  __shared__ half_t Bs[TN][TKB + 8];   // transposed: Bs[n][k]

  const int z = blockIdx.z;
  const long long offA = (long long)(z >> 2) * sAb + (long long)(z & 3) * sAh;
  const long long offB = (long long)(z >> 2) * sBb + (long long)(z & 3) * sBh;
  const long long offC = (long long)(z >> 2) * sCb + (long long)(z & 3) * sCh;

  const int tid = threadIdx.x;
  const int lane = tid & 31;
  const int wid = tid >> 5;     // 0..7
  const int wm = wid >> 1;      // 0..3 : 32-row band
  const int wn = wid & 1;       // 0..1 : 32-col band
  const int m0 = blockIdx.x * TM;
  const int n0 = blockIdx.y * TN;

  const int half_id = lane >> 4;
  const int l16 = lane & 15;

  v8f acc[2][2] = {};

  for (int ks = 0; ks < K; ks += TKB) {
    __syncthreads();   // previous tiles fully consumed

    // ---- A tile [128][64] halves = 16KB: 1024 x 16B chunks via async DMA ----
#pragma unroll
    for (int kc = 0; kc < 4; ++kc) {
      const int c = tid + kc * 256;        // 0..1023
      const int r = c >> 3;                // 0..127
      const int ch = (c & 7) * 8;          // half offset 0..56
      // low 32 bits of the flat address of an LDS object == LDS byte offset
      const unsigned lds = (unsigned)(unsigned long long)(void*)&As[r][ch];
      const half_t* g = A + offA + (size_t)(m0 + r) * lda + ks + ch;
      asm volatile("global_load_async_to_lds_b128 %0, %1, off"
                   :: "v"(lds), "v"(g) : "memory");
    }

    // ---- B tile [64][64] halves, stored transposed Bs[n][k] ----
#pragma unroll
    for (int kc = 0; kc < 2; ++kc) {
      const int c = tid + kc * 256;        // 0..511
      const int kk = c >> 3;               // 0..63
      const int nc = (c & 7) * 8;          // 0..56
      v8h bv = *(const v8h*)(B + offB + (size_t)(ks + kk) * ldb + n0 + nc);
#pragma unroll
      for (int i = 0; i < 8; ++i) Bs[nc + i][kk] = bv[i];
    }

    asm volatile("s_wait_asynccnt 0x0" ::: "memory");  // A tile landed in LDS
    __syncthreads();

    // ---- 2 x (2x2) WMMA: 8 v_wmma per staged tile ----
#pragma unroll
    for (int kb = 0; kb < 2; ++kb) {
      const int ko = kb * 32;
      v16h afr[2], bfr[2];
#pragma unroll
      for (int sm = 0; sm < 2; ++sm) {
        const int mr = wm * 32 + sm * 16 + l16;
        v8h lo = *(const v8h*)&As[mr][ko + half_id * 8];        // K: half*8 .. +7
        v8h hi = *(const v8h*)&As[mr][ko + 16 + half_id * 8];   // K: 16+half*8 .. +7
#pragma unroll
        for (int i = 0; i < 8; ++i) { afr[sm][i] = lo[i]; afr[sm][8 + i] = hi[i]; }
      }
#pragma unroll
      for (int sn = 0; sn < 2; ++sn) {
        const int nr = wn * 32 + sn * 16 + l16;
        v8h lo = *(const v8h*)&Bs[nr][ko + half_id * 8];        // K: half*8 .. +7
        v8h hi = *(const v8h*)&Bs[nr][ko + 16 + half_id * 8];   // K: 16+half*8 .. +7
#pragma unroll
        for (int i = 0; i < 8; ++i) { bfr[sn][i] = lo[i]; bfr[sn][8 + i] = hi[i]; }
      }
#pragma unroll
      for (int sm = 0; sm < 2; ++sm)
#pragma unroll
        for (int sn = 0; sn < 2; ++sn)
          acc[sm][sn] = __builtin_amdgcn_wmma_f32_16x16x32_f16(
              false, afr[sm], false, bfr[sn], (short)0, acc[sm][sn], false, false);
    }
  }

  // epilogue: D[m][n]; lane l16 = col, row = v + 8*half
#pragma unroll
  for (int sm = 0; sm < 2; ++sm) {
#pragma unroll
    for (int sn = 0; sn < 2; ++sn) {
      const int gn = n0 + wn * 32 + sn * 16 + l16;
      const float bv = bias ? bias[gn] : 0.0f;
#pragma unroll
      for (int v = 0; v < 8; ++v) {
        const int gm = m0 + wm * 32 + sm * 16 + half_id * 8 + v;
        float x = acc[sm][sn][v] * scale + bv;
        if (do_gelu) x = gelu_f(x);
        const long long ci = offC + (long long)gm * ldc + gn;
        if (C32) C32[ci] = x;
        if (C16) C16[ci] = (half_t)x;
      }
    }
  }
}

// ---------------- host-side orchestration ----------------

extern "C" void kernel_launch(void* const* d_in, const int* in_sizes, int n_in,
                              void* d_out, int out_size, void* d_ws, size_t ws_size,
                              hipStream_t stream) {
  const float* h_f         = (const float*)d_in[0];
  const float* input_pos   = (const float*)d_in[1];
  const float* prop_pos    = (const float*)d_in[2];
  const float* b_ff        = (const float*)d_in[3];
  const float* cp_w1       = (const float*)d_in[4];
  const float* cp_w2       = (const float*)d_in[5];
  const float* ln1_g       = (const float*)d_in[6];
  const float* ln1_b       = (const float*)d_in[7];
  const float* ln2_g       = (const float*)d_in[8];
  const float* ln2_b       = (const float*)d_in[9];
  const float* wq          = (const float*)d_in[10];
  const float* wk          = (const float*)d_in[11];
  const float* wv          = (const float*)d_in[12];
  const float* wo          = (const float*)d_in[13];
  const float* bo          = (const float*)d_in[14];
  const float* ffn_w1      = (const float*)d_in[15];
  const float* ffn_b1      = (const float*)d_in[16];
  const float* ffn_w2      = (const float*)d_in[17];
  const float* ffn_b2      = (const float*)d_in[18];
  const float* dec_ln_g    = (const float*)d_in[19];
  const float* dec_ln_b    = (const float*)d_in[20];
  const float* dec_w1      = (const float*)d_in[21];
  const float* dec_w2      = (const float*)d_in[22];

  char* W = (char*)d_ws;
  const size_t MB = 1ull << 20;
  // f16 weight pool
  half_t* w_cpw1 = (half_t*)(W + 0);
  half_t* w_cpw2 = (half_t*)(W + 32 * 1024);
  half_t* w_wq   = (half_t*)(W + 64 * 1024);
  half_t* w_wk   = (half_t*)(W + 192 * 1024);
  half_t* w_wv   = (half_t*)(W + 320 * 1024);
  half_t* w_wo   = (half_t*)(W + 448 * 1024);
  half_t* w_fw1  = (half_t*)(W + 576 * 1024);
  half_t* w_fw2  = (half_t*)(W + 608 * 1024);
  half_t* w_dw1  = (half_t*)(W + 640 * 1024);
  // activations (aliased by lifetime)
  float*  x_f32   = (float*)(W + 1 * MB);            // 32MB  (coord-proj residual)
  float*  x2_f32  = (float*)(W + 33 * MB);           // 32MB
  half_t* S5      = (half_t*)(W + 65 * MB);          // 17MB: x_h -> x2_h -> hd_h
  half_t* z_h     = (half_t*)(W + 82 * MB);          // 5MB
  half_t* S6      = (half_t*)(W + 87 * MB);          // 17MB: k_h -> kT_h -> f1_h
  half_t* S7      = (half_t*)(W + 104 * MB);         // 17MB: k_r -> v_h
  half_t* S8      = (half_t*)(W + 121 * MB);         // 17MB: vp_h -> d1_h
  half_t* dots_h  = (half_t*)(W + 138 * MB);         // 1MB
  half_t* S1      = (half_t*)(W + 139 * MB);         // 68MB: ff_h/t1_h -> q_h -> out_h
  half_t* S2h     = (half_t*)(W + 207 * MB);         // 68MB: q_p -> attn_f32/ffn_f32
  float*  S2f     = (float*)(W + 207 * MB);

  half_t* ff_h  = S1;
  half_t* t1_h  = (half_t*)((char*)S1 + 17 * MB);
  half_t* q_h   = S1;
  half_t* out_h = S1;
  half_t* x_h   = S5;
  half_t* x2_h  = S5;
  half_t* hd_h  = S5;
  half_t* k_h   = S6;
  half_t* kT_h  = S6;
  half_t* f1_h  = S6;
  half_t* k_r   = S7;
  half_t* v_h   = S7;
  half_t* vp_h  = S8;
  half_t* d1_h  = S8;
  half_t* q_p   = S2h;
  float*  attn_f32 = S2f;
  float*  ffn_f32  = (float*)((char*)S2f + 34 * MB);

  auto cvt = [&](const float* s, half_t* d, int n) {
    k_cvt_f32_f16<<<(n + 255) / 256, 256, 0, stream>>>(s, d, n);
  };
  auto gemm = [&](const half_t* A, const half_t* B, float* C32, half_t* C16,
                  const float* bias, int M, int N, int K, int lda, int ldb, int ldc,
                  long long sAb, long long sAh, long long sBb, long long sBh,
                  long long sCb, long long sCh, float scale, int do_gelu, int batch) {
    dim3 g(M / TM, N / TN, batch);
    k_wmma_gemm<<<g, 256, 0, stream>>>(A, B, C32, C16, bias, M, N, K, lda, ldb, ldc,
                                       sAb, sAh, sBb, sBh, sCb, sCh, scale, do_gelu);
  };

  // 0) weights -> f16
  cvt(cp_w1, w_cpw1, 128 * 128);
  cvt(cp_w2, w_cpw2, 128 * 128);
  cvt(wq, w_wq, 128 * 512);
  cvt(wk, w_wk, 128 * 512);
  cvt(wv, w_wv, 128 * 512);
  cvt(wo, w_wo, 512 * 128);
  cvt(ffn_w1, w_fw1, 128 * 128);
  cvt(ffn_w2, w_fw2, 128 * 128);
  cvt(dec_w1, w_dw1, 128 * 64);

  // 1) Fourier features for query points
  k_fourier<<<(NTOKQ * 64 + 255) / 256, 256, 0, stream>>>(prop_pos, b_ff, ff_h, NTOKQ);

  // 2) coord MLP: t1 = gelu(ff @ cp_w1);  x = t1 @ cp_w2
  gemm(ff_h, w_cpw1, nullptr, t1_h, nullptr, NTOKQ, 128, 128, 128, 128, 128,
       0, 0, 0, 0, 0, 0, 1.0f, 1, 1);
  gemm(t1_h, w_cpw2, x_f32, x_h, nullptr, NTOKQ, 128, 128, 128, 128, 128,
       0, 0, 0, 0, 0, 0, 1.0f, 0, 1);

  // 3) z = LN(h)
  k_ln128<<<NTOKIN, 128, 0, stream>>>(h_f, ln1_g, ln1_b, z_h);

  // 4) projections
  gemm(x_h, w_wq, nullptr, q_h, nullptr, NTOKQ, 512, 128, 128, 512, 512,
       0, 0, 0, 0, 0, 0, 1.0f, 0, 1);
  k_rope_q<<<(NTOKQ * 512 + 255) / 256, 256, 0, stream>>>(q_h, prop_pos, q_p);

  gemm(z_h, w_wk, nullptr, k_h, nullptr, NTOKIN, 512, 128, 128, 512, 512,
       0, 0, 0, 0, 0, 0, 1.0f, 0, 1);
  k_rope_k<<<(NTOKIN * 512 + 255) / 256, 256, 0, stream>>>(k_h, input_pos, k_r);
  k_inorm_kT<<<B_ * 512, 256, 0, stream>>>(k_r, kT_h, NIN);   // -> [z][d][n]

  gemm(z_h, w_wv, nullptr, v_h, nullptr, NTOKIN, 512, 128, 128, 512, 512,
       0, 0, 0, 0, 0, 0, 1.0f, 0, 1);
  k_inorm_v<<<B_ * 512, 256, 0, stream>>>(v_h, vp_h, NIN);    // -> [z][n][d]

  // 5) Galerkin attention: dots = (k^T v)/n_in ; out = q @ dots
  {
    const long long sK = (long long)128 * NIN;   // kT per-z stride
    const long long sV = (long long)NIN * 128;   // vp per-z stride
    const long long sD = 128 * 128;              // dots per-z stride
    gemm(kT_h, vp_h, nullptr, dots_h, nullptr, 128, 128, NIN, NIN, 128, 128,
         4 * sK, sK, 4 * sV, sV, 4 * sD, sD, 1.0f / (float)NIN, 0, 16);
    const long long sQ = (long long)NQ * 128;    // q_p per-z stride
    gemm(q_p, dots_h, nullptr, out_h, nullptr, NQ, 128, 128, 128, 128, 512,
         4 * sQ, sQ, 4 * sD, sD, (long long)NQ * 512, 128, 1.0f, 0, 16);
  }

  // 6) attn = out @ wo + bo
  gemm(out_h, w_wo, attn_f32, nullptr, bo, NTOKQ, 128, 512, 512, 128, 128,
       0, 0, 0, 0, 0, 0, 1.0f, 0, 1);

  // 7) x2 = LN(attn) + x
  k_ln128_res<<<NTOKQ, 128, 0, stream>>>(attn_f32, x_f32, ln2_g, ln2_b, x2_f32, x2_h);

  // 8) FFN
  gemm(x2_h, w_fw1, nullptr, f1_h, ffn_b1, NTOKQ, 128, 128, 128, 128, 128,
       0, 0, 0, 0, 0, 0, 1.0f, 1, 1);
  gemm(f1_h, w_fw2, ffn_f32, nullptr, ffn_b2, NTOKQ, 128, 128, 128, 128, 128,
       0, 0, 0, 0, 0, 0, 1.0f, 0, 1);

  // 9) hd = LN(ffn + x2)
  k_addln128<<<NTOKQ, 128, 0, stream>>>(ffn_f32, x2_f32, dec_ln_g, dec_ln_b, hd_h);

  // 10) decode: d1 = gelu(hd @ dec_w1);  out = d1 @ dec_w2
  gemm(hd_h, w_dw1, nullptr, d1_h, nullptr, NTOKQ, 64, 128, 128, 64, 64,
       0, 0, 0, 0, 0, 0, 1.0f, 1, 1);
  k_decode<<<(NTOKQ + 255) / 256, 256, 0, stream>>>(d1_h, dec_w2, (float*)d_out, NTOKQ);
}
